// AttentionShareGlobal_8461085573552
// MI455X (gfx1250) — matrix operand order repaired and checked by reading, not compiled, not run on hardware
//
#include <hip/hip_runtime.h>
#include <hip/hip_bf16.h>

typedef __attribute__((ext_vector_type(16))) _Float16 v16h;
typedef __attribute__((ext_vector_type(4)))  _Float16 v4h;
typedef __attribute__((ext_vector_type(8)))  float    v8f;

#define Bx   256
#define Nn   196
#define DIM  256
#define NH   8
#define HD   32
#define G    14
#define NT   13            // 13 tiles of 16 cover 208 >= 196
#define NPAD 208           // padded N/M for the bias table and P columns
#define KPAD 224           // 7 K-chunks of 32 for P@V

// single-instruction lane-xor exchange via ds_swizzle_b32 (group-of-32 mode:
// offset = xor<<10 | or<<5 | and; and=0x1f, or=0). Masks < 16 stay inside each
// 16-lane half, matching the WMMA C-layout row split.
#define SWZ_XOR(x, m) \
  __int_as_float(__builtin_amdgcn_ds_swizzle(__float_as_int(x), ((m) << 10) | 0x1f))

// ---- WMMA fragment K-index patterns (wave32, 16-bit operands) ----
// A-matrix 16x32 (ISA 7.12.2): lanes 0-15 hold K={0..7,16..23}, lanes 16-31 K={8..15,24..31}
__device__ __forceinline__ int kpatA(int lane, int v) {
  int lo = (lane < 16) ? 0 : 8;
  return (v < 4) ? (lo + 2 * v) : (lo + 16 + 2 * (v - 4));
}
// B-matrix 32x16: lane = column, same K pattern as A: lanes 0-15 K={0..7,16..23}, lanes 16-31 K={8..15,24..31}
__device__ __forceinline__ int kpatB(int lane, int v) {
  int lo = (lane < 16) ? 0 : 8;
  return (v < 4) ? (lo + 2 * v) : (lo + 16 + 2 * (v - 4));
}

// ---------------- Kernel 1: DPB MLP  pos[729][8] ----------------
__global__ void dpb_mlp_kernel(const float* __restrict__ w1, const float* __restrict__ b1,
                               const float* __restrict__ w2, const float* __restrict__ b2,
                               float* __restrict__ pos) {
  int t = blockIdx.x * blockDim.x + threadIdx.x;
  if (t >= 729) return;                       // (2g-1)^2 = 27*27
  float c0 = (float)(t / 27 - (G - 1));
  float c1 = (float)(t % 27 - (G - 1));
  float o[NH];
  #pragma unroll
  for (int j = 0; j < NH; ++j) o[j] = b2[j];
  for (int kk = 0; kk < 64; ++kk) {
    float h = fmaxf(0.f, fmaf(c0, w1[kk], fmaf(c1, w1[64 + kk], b1[kk])));
    #pragma unroll
    for (int j = 0; j < NH; ++j) o[j] = fmaf(h, w2[kk * NH + j], o[j]);
  }
  #pragma unroll
  for (int j = 0; j < NH; ++j) pos[t * NH + j] = o[j];
}

// ------- Kernel 2: gather padded bias table rpb[H][208][208] -------
// cols >=196 carry -1e30 (column softmax mask, S there is 0 from zero-padded K);
// rows >=196 carry 0 (those output rows are never stored).
__global__ void rpb_gather_kernel(const float* __restrict__ pos, float* __restrict__ rpb) {
  int t = blockIdx.x * blockDim.x + threadIdx.x;
  if (t >= NPAD * NPAD) return;
  int i = t / NPAD, j = t % NPAD;
  if (j >= Nn) {
    #pragma unroll
    for (int hh = 0; hh < NH; ++hh)
      rpb[(size_t)hh * (NPAD * NPAD) + t] = -1.0e30f;
  } else if (i >= Nn) {
    #pragma unroll
    for (int hh = 0; hh < NH; ++hh)
      rpb[(size_t)hh * (NPAD * NPAD) + t] = 0.0f;
  } else {
    int ci = i / G, di = i % G, cj = j / G, dj = j % G;
    int idx = (ci - cj + (G - 1)) * (2 * G - 1) + (di - dj + (G - 1));
    #pragma unroll
    for (int hh = 0; hh < NH; ++hh)
      rpb[(size_t)hh * (NPAD * NPAD) + t] = pos[idx * NH + hh];
  }
}

// ---------------- Kernel 3: fused flash-style attention ----------------
// grid = (13 row tiles, NH, B), block = one wave32
__global__ __launch_bounds__(32)
void attn_kernel(const float* __restrict__ q, const float* __restrict__ k,
                 const float* __restrict__ v, const float* __restrict__ rpb,
                 _Float16* __restrict__ xws) {
  const int tile = blockIdx.x, h = blockIdx.y, b = blockIdx.z;
  const int lane = threadIdx.x;
  const int lg = lane & 15, hihalf = lane >> 4;
  const int rb = tile * 16;

  __shared__ _Float16 Ql[16][HD];        //  1 KB
  __shared__ _Float16 Kl[KPAD][HD];      // 14 KB (rows >=196 zero)
  __shared__ _Float16 Vl[KPAD][HD];      // 14 KB (rows >=196 zero)
  __shared__ _Float16 Pl[16][KPAD];      //  7 KB

  const float scale = 0.17677669529663687f;   // 32^-0.5, folded into Q

  // ---- vectorized staging: global f32x4 -> LDS f16x4 ----
  for (int idx = lane; idx < 16 * (HD / 4); idx += 32) {
    int m = idx >> 3, d = (idx & 7) * 4;
    int row = rb + m;
    float4 qv = make_float4(0.f, 0.f, 0.f, 0.f);
    if (row < Nn)
      qv = *(const float4*)(q + ((size_t)(b * Nn + row)) * DIM + h * HD + d);
    v4h qh4 = {(_Float16)(qv.x * scale), (_Float16)(qv.y * scale),
               (_Float16)(qv.z * scale), (_Float16)(qv.w * scale)};
    *(v4h*)&Ql[m][d] = qh4;
  }
  for (int idx = lane; idx < KPAD * (HD / 4); idx += 32) {
    int m = idx >> 3, d = (idx & 7) * 4;
    float4 kv = make_float4(0.f, 0.f, 0.f, 0.f), vv = kv;
    if (m < Nn) {
      const size_t base = ((size_t)(b * Nn + m)) * DIM + h * HD + d;
      kv = *(const float4*)(k + base);
      vv = *(const float4*)(v + base);
    }
    v4h kh4 = {(_Float16)kv.x, (_Float16)kv.y, (_Float16)kv.z, (_Float16)kv.w};
    v4h vh4 = {(_Float16)vv.x, (_Float16)vv.y, (_Float16)vv.z, (_Float16)vv.w};
    *(v4h*)&Kl[m][d] = kh4;
    *(v4h*)&Vl[m][d] = vh4;
  }
  // zero the P pad columns 208..223 (exp of -1e30 handles 196..207)
  for (int idx = lane; idx < 16 * 4; idx += 32)
    *(v4h*)&Pl[idx >> 2][NPAD + (idx & 3) * 4] = (v4h){(_Float16)0.f, (_Float16)0.f,
                                                       (_Float16)0.f, (_Float16)0.f};
  __syncthreads();

  // ---- S = (Q*scale) @ K^T : 13 WMMAs, all in registers ----
  v16h aq;
  #pragma unroll
  for (int vv = 0; vv < 8; ++vv) {
    int k0 = kpatA(lane, vv);
    aq[2 * vv]     = Ql[lg][k0];
    aq[2 * vv + 1] = Ql[lg][k0 + 1];
  }
  v8f acc[NT];
  #pragma unroll
  for (int t = 0; t < NT; ++t) {
    v16h bk;
    int jrow = t * 16 + lg;               // key index (column of S)
    #pragma unroll
    for (int vv = 0; vv < 8; ++vv) {
      int k0 = kpatB(lane, vv);
      bk[2 * vv]     = Kl[jrow][k0];
      bk[2 * vv + 1] = Kl[jrow][k0 + 1];
    }
    v8f c = {};
    acc[t] = __builtin_amdgcn_wmma_f32_16x16x32_f16(false, aq, false, bk,
                                                    (short)0, c, false, false);
  }

  // ---- branch-free bias add: one base address, immediate offsets ----
  // element (r, lane) holds S[row = rb + r + 8*hihalf][col = t*16 + lg]
  {
    const float* bbase =
        rpb + ((size_t)h * NPAD + rb + hihalf * 8) * NPAD + lg;
    #pragma unroll
    for (int r = 0; r < 8; ++r)
      #pragma unroll
      for (int t = 0; t < NT; ++t)
        acc[t][r] += bbase[r * NPAD + t * 16];
  }

  // ---- row softmax: 13-reg serial reduce + ds_swizzle xor-butterflies ----
  #pragma unroll
  for (int r = 0; r < 8; ++r) {
    float mx = acc[0][r];
    #pragma unroll
    for (int t = 1; t < NT; ++t) mx = fmaxf(mx, acc[t][r]);
    mx = fmaxf(mx, SWZ_XOR(mx, 1));
    mx = fmaxf(mx, SWZ_XOR(mx, 2));
    mx = fmaxf(mx, SWZ_XOR(mx, 4));
    mx = fmaxf(mx, SWZ_XOR(mx, 8));
    float sum = 0.f;
    #pragma unroll
    for (int t = 0; t < NT; ++t) {
      float e = __expf(acc[t][r] - mx);
      acc[t][r] = e; sum += e;
    }
    sum += SWZ_XOR(sum, 1);
    sum += SWZ_XOR(sum, 2);
    sum += SWZ_XOR(sum, 4);
    sum += SWZ_XOR(sum, 8);
    float inv = __frcp_rn(sum);
    #pragma unroll
    for (int t = 0; t < NT; ++t) acc[t][r] *= inv;
  }

  // ---- C->A relayout through LDS ----
  #pragma unroll
  for (int t = 0; t < NT; ++t)
    #pragma unroll
    for (int r = 0; r < 8; ++r)
      Pl[r + hihalf * 8][t * 16 + lg] = (_Float16)acc[t][r];
  __syncthreads();

  // ---- X = P @ V : 7 K-chunks x 2 N-tiles ----
  v8f x0 = {}, x1 = {};
  #pragma unroll
  for (int kc = 0; kc < 7; ++kc) {
    v16h ap;
    #pragma unroll
    for (int vv = 0; vv < 8; ++vv) {
      int k0 = kc * 32 + kpatA(lane, vv);
      ap[2 * vv]     = Pl[lg][k0];
      ap[2 * vv + 1] = Pl[lg][k0 + 1];
    }
    v16h bv0, bv1;
    #pragma unroll
    for (int vv = 0; vv < 8; ++vv) {
      int kk = kc * 32 + kpatB(lane, vv);
      bv0[2 * vv]     = Vl[kk][lg];
      bv0[2 * vv + 1] = Vl[kk + 1][lg];
      bv1[2 * vv]     = Vl[kk][16 + lg];
      bv1[2 * vv + 1] = Vl[kk + 1][16 + lg];
    }
    x0 = __builtin_amdgcn_wmma_f32_16x16x32_f16(false, ap, false, bv0, (short)0, x0, false, false);
    x1 = __builtin_amdgcn_wmma_f32_16x16x32_f16(false, ap, false, bv1, (short)0, x1, false, false);
  }

  // ---- store head output (f16) in [B, N, H*HD] layout for projection ----
  #pragma unroll
  for (int r = 0; r < 8; ++r) {
    int i = rb + r + hihalf * 8;
    if (i < Nn) {
      _Float16* xr = xws + ((size_t)(b * Nn + i)) * DIM + h * HD;
      xr[lg]      = (_Float16)x0[r];
      xr[16 + lg] = (_Float16)x1[r];
    }
  }
}

// ---------------- Kernel 4: output projection GEMM ----------------
// x[50176,256](f16) @ W[256,256] + b -> out (f32). 4 waves/block, wave = 16x64 tile.
__global__ __launch_bounds__(128)
void proj_kernel(const _Float16* __restrict__ xws, const float* __restrict__ w,
                 const float* __restrict__ bias, float* __restrict__ out) {
  const int mt = blockIdx.x;
  const int wave = threadIdx.x >> 5;
  const int lane = threadIdx.x & 31;
  const int lg = lane & 15, hihalf = lane >> 4;
  const int mbase = mt * 16;
  const int cbase = wave * 64;

  v8f acc0 = {}, acc1 = {}, acc2 = {}, acc3 = {};
  for (int kc = 0; kc < 8; ++kc) {
    v16h a;
    const _Float16* xr = xws + (size_t)(mbase + lg) * DIM + kc * 32;
    #pragma unroll
    for (int vv = 0; vv < 8; ++vv) {
      int k0 = kpatA(lane, vv);
      a[2 * vv] = xr[k0]; a[2 * vv + 1] = xr[k0 + 1];
    }
    #pragma unroll
    for (int nt = 0; nt < 4; ++nt) {
      v16h bm;
      int col = cbase + nt * 16 + lg;
      #pragma unroll
      for (int vv = 0; vv < 8; ++vv) {
        int kk = kc * 32 + kpatB(lane, vv);
        bm[2 * vv]     = (_Float16)w[(size_t)kk * DIM + col];
        bm[2 * vv + 1] = (_Float16)w[(size_t)(kk + 1) * DIM + col];
      }
      v8f* accp = (nt == 0) ? &acc0 : (nt == 1) ? &acc1 : (nt == 2) ? &acc2 : &acc3;
      *accp = __builtin_amdgcn_wmma_f32_16x16x32_f16(false, a, false, bm,
                                                     (short)0, *accp, false, false);
    }
  }
  v8f accs[4] = {acc0, acc1, acc2, acc3};
  #pragma unroll
  for (int nt = 0; nt < 4; ++nt) {
    int col = cbase + nt * 16 + lg;
    float bb = bias[col];
    #pragma unroll
    for (int r = 0; r < 8; ++r) {
      int row = mbase + r + hihalf * 8;
      out[(size_t)row * DIM + col] = accs[nt][r] + bb;
    }
  }
}

extern "C" void kernel_launch(void* const* d_in, const int* in_sizes, int n_in,
                              void* d_out, int out_size, void* d_ws, size_t ws_size,
                              hipStream_t stream) {
  const float* q  = (const float*)d_in[0];
  const float* k  = (const float*)d_in[1];
  const float* v  = (const float*)d_in[2];
  const float* w1 = (const float*)d_in[3];
  const float* b1 = (const float*)d_in[4];
  const float* w2 = (const float*)d_in[5];
  const float* b2 = (const float*)d_in[6];
  const float* pw = (const float*)d_in[7];
  const float* pb = (const float*)d_in[8];
  (void)in_sizes; (void)n_in; (void)out_size; (void)ws_size;

  // workspace: pos[729*8] f32 | rpb[8*208*208] f32 (padded+masked) | x[50176*256] f16
  float* pos = (float*)d_ws;
  float* rpb = (float*)((char*)d_ws + 24576);
  _Float16* xws =
      (_Float16*)((char*)d_ws + 24576 + (size_t)NH * NPAD * NPAD * 4);

  dpb_mlp_kernel<<<3, 256, 0, stream>>>(w1, b1, w2, b2, pos);
  rpb_gather_kernel<<<(NPAD * NPAD) / 256, 256, 0, stream>>>(pos, rpb);
  attn_kernel<<<dim3(NT, NH, Bx), 32, 0, stream>>>(q, k, v, rpb, xws);
  proj_kernel<<<(Bx * Nn) / 16, 128, 0, stream>>>(xws, pw, pb, (float*)d_out);
}
